// FlaxMoE_42880953483997
// MI455X (gfx1250) — matrix operand order repaired and checked by reading, not compiled, not run on hardware
//
#include <hip/hip_runtime.h>
#include <hip/hip_bf16.h>
#include <math.h>

// ---------------- model dims ----------------
#define T_TOK   2048
#define DMODEL  512
#define HDIM    1024
#define NEXP    8
#define TWO_H   2048

// ---------------- tiling ----------------
#define ROWS_PER_TILE 32          // M rows per workgroup
#define TILES_MAX     64          // 64*32 = 2048 covers worst-case expert load
#define SA_STRIDE     (DMODEL + 8)   // 520 bf16 -> 1040B row (16B aligned)
#define SH_COLS       128            // h-chunk width
#define SH_STRIDE     (SH_COLS + 8)  // 136 bf16 -> 272B row (16B aligned)

typedef __attribute__((ext_vector_type(16))) __bf16 v16bf;
typedef __attribute__((ext_vector_type(8)))  __bf16 v8bf;
typedef __attribute__((ext_vector_type(8)))  float  v8f;
typedef __attribute__((ext_vector_type(4)))  float  v4f;

// ---------------------------------------------------------------------------
// A-fragment (16x32 bf16) from LDS, per CDNA5 ISA 7.12.2 16-bit A layout:
//   lane L: row = L&15 ; VGPR v holds K pair {base+2(v&3), +1},
//   base = (v>=4 ? 16 : 0) + (L>=16 ? 8 : 0)
// -> two contiguous 8-element (b128) LDS reads per lane.
// ---------------------------------------------------------------------------
static __device__ inline v16bf load_a_frag(const __bf16* lds, int rowBase,
                                           int stride, int kBase, int lane) {
  const int r  = rowBase + (lane & 15);
  const int kh = (lane >= 16) ? 8 : 0;
  const __bf16* p = lds + r * stride + kBase + kh;
  v8bf lo = *(const v8bf*)(p);        // K = kBase+kh   .. +8
  v8bf hi = *(const v8bf*)(p + 16);   // K = kBase+16+kh .. +8
  return __builtin_shufflevector(lo, hi,
                                 0,1,2,3,4,5,6,7,8,9,10,11,12,13,14,15);
}

// ---------------------------------------------------------------------------
// B-fragment (32x16 bf16): lane L holds col N=L&15; element s holds
// K = 8*(L>=16) + s (s<8), 16 + 8*(L>=16) + (s-8) (s>=8). Source is fp32
// row-major in global memory (row stride `stride`), converted to bf16.
// ---------------------------------------------------------------------------
static __device__ inline v16bf load_b_frag(const float* __restrict__ p, int stride, int lane) {
  const float* q = p + (lane & 15) + (size_t)((lane >= 16) ? 8 : 0) * stride;
  v16bf b;
#pragma unroll
  for (int s = 0; s < 8; ++s) {
    b[s]     = (__bf16)q[(size_t)s * stride];
    b[8 + s] = (__bf16)q[(size_t)(16 + s) * stride];
  }
  return b;
}

#define WMMA_BF16(A, B, C) \
  __builtin_amdgcn_wmma_f32_16x16x32_bf16(false, (A), false, (B), (short)0, (C), false, false)

// ===========================================================================
// Router: logits, top-2 gates, per-token softmax stats, scatter to lists.
// ===========================================================================
__global__ __launch_bounds__(256)
void router_kernel(const float* __restrict__ x,
                   const float* __restrict__ w_router,
                   float* __restrict__ gatesOut,   // [T*2]
                   int*   __restrict__ counts,     // [E]
                   int*   __restrict__ lists,      // [E][T] -> t*2+k
                   float* __restrict__ probsBuf,   // [T][E]
                   float* __restrict__ lseSqBuf) { // [T]
  const int t = blockIdx.x * blockDim.x + threadIdx.x;
  if (t >= T_TOK) return;

  float lg[NEXP];
#pragma unroll
  for (int e = 0; e < NEXP; ++e) lg[e] = 0.f;
  const float* xr = x + (size_t)t * DMODEL;
  for (int d = 0; d < DMODEL; ++d) {
    const float xv = xr[d];
#pragma unroll
    for (int e = 0; e < NEXP; ++e) lg[e] += xv * w_router[d * NEXP + e];
  }

  // top-2 (ties -> lowest index, matching jax.lax.top_k)
  int i0 = 0;
#pragma unroll
  for (int e = 1; e < NEXP; ++e) if (lg[e] > lg[i0]) i0 = e;
  int i1 = (i0 == 0) ? 1 : 0;
#pragma unroll
  for (int e = 0; e < NEXP; ++e)
    if (e != i0 && lg[e] > lg[i1]) i1 = e;

  const float z  = expf(lg[i1] - lg[i0]);   // lg[i0] >= lg[i1]
  const float g0 = 1.f / (1.f + z);
  const float g1 = z * g0;
  gatesOut[t * 2 + 0] = g0;
  gatesOut[t * 2 + 1] = g1;

  // full softmax + logsumexp^2 (for aux loss)
  const float mx = lg[i0];
  float ex[NEXP], s = 0.f;
#pragma unroll
  for (int e = 0; e < NEXP; ++e) { ex[e] = expf(lg[e] - mx); s += ex[e]; }
  const float inv = 1.f / s;
#pragma unroll
  for (int e = 0; e < NEXP; ++e) probsBuf[t * NEXP + e] = ex[e] * inv;
  const float lse = mx + logf(s);
  lseSqBuf[t] = lse * lse;

  // scatter (slot order nondeterminism is harmless: rows are independent and
  // each (t,k) writes a fixed output address)
  const int s0 = atomicAdd(&counts[i0], 1);
  lists[i0 * T_TOK + s0] = t * 2 + 0;
  const int s1 = atomicAdd(&counts[i1], 1);
  lists[i1 * T_TOK + s1] = t * 2 + 1;
}

// ===========================================================================
// Fused per-expert FFN: (GEMM1 -> SiLU*gate -> GEMM2) with bf16 WMMA.
// One WG = 8 waves = one (expert, 32-row) tile. C = [32 x 512] fp32 in regs.
// ===========================================================================
__global__ __launch_bounds__(256)
void moe_ffn_kernel(const float* __restrict__ x,
                    const float* __restrict__ w_in,   // [E][D][2H]
                    const float* __restrict__ w_out,  // [E][H][D]
                    const int*   __restrict__ counts,
                    const int*   __restrict__ lists,
                    const float* __restrict__ gates,
                    float*       __restrict__ ybuf) { // [2][T][D]
  __shared__ __attribute__((aligned(16))) __bf16 sA[ROWS_PER_TILE * SA_STRIDE];
  __shared__ __attribute__((aligned(16))) __bf16 sH[ROWS_PER_TILE * SH_STRIDE];
  __shared__ int   sTok[ROWS_PER_TILE];
  __shared__ float sGate[ROWS_PER_TILE];

  const int e  = blockIdx.x >> 6;       // / TILES_MAX
  const int mt = blockIdx.x & (TILES_MAX - 1);
  const int ne = counts[e];
  const int m0 = mt * ROWS_PER_TILE;
  if (m0 >= ne) return;                 // uniform across the block

  const int tid  = threadIdx.x;
  const int lane = tid & 31;
  const int wid  = tid >> 5;            // 8 waves

  const float* wInE  = w_in  + (size_t)e * DMODEL * TWO_H;
  const float* wOutE = w_out + (size_t)e * HDIM * DMODEL;

  // ---- row metadata ----
  if (tid < ROWS_PER_TILE) {
    const int gm = m0 + tid;
    const int entry = (gm < ne) ? lists[e * T_TOK + gm] : -1;
    sTok[tid]  = entry;
    sGate[tid] = (entry >= 0) ? gates[entry] : 0.f;
  }
  __syncthreads();

  // ---- stage A rows (x, fp32 -> bf16) into LDS ----
  {
    const int r  = tid >> 3;            // 0..31
    const int c0 = (tid & 7) * 64;      // 0..448
    const int entry = sTok[r];
    __bf16* dst = &sA[r * SA_STRIDE + c0];
    if (entry >= 0) {
      const float* xrow = x + (size_t)(entry >> 1) * DMODEL + c0;
#pragma unroll
      for (int c = 0; c < 64; c += 4) {
        v4f f = *(const v4f*)(xrow + c);
        dst[c + 0] = (__bf16)f.x; dst[c + 1] = (__bf16)f.y;
        dst[c + 2] = (__bf16)f.z; dst[c + 3] = (__bf16)f.w;
      }
    } else {
#pragma unroll
      for (int c = 0; c < 64; ++c) dst[c] = (__bf16)0.f;
    }
  }
  __syncthreads();

  const int hwBase = wid * 16;   // this wave's h-cols within the chunk
  const int nBase  = wid * 64;   // this wave's D-cols of the final output

  v8f accC[2][4];
#pragma unroll
  for (int i = 0; i < 2; ++i)
#pragma unroll
    for (int n = 0; n < 4; ++n)
#pragma unroll
      for (int q = 0; q < 8; ++q) accC[i][n][q] = 0.f;

  for (int ch = 0; ch < HDIM / SH_COLS; ++ch) {   // 8 chunks of 128 h-cols
    const int hGlob = ch * SH_COLS + hwBase;

    // hint the GEMM2 B rows for this chunk into cache
    __builtin_prefetch(wOutE + (size_t)(ch * SH_COLS + lane) * DMODEL + nBase, 0, 1);

    // ---------------- GEMM1: [32 x 512] x [512 x 16] (act & gate halves) ---
    v8f accA[2], accG[2];
#pragma unroll
    for (int i = 0; i < 2; ++i)
#pragma unroll
      for (int q = 0; q < 8; ++q) { accA[i][q] = 0.f; accG[i][q] = 0.f; }

    for (int kk = 0; kk < DMODEL / 32; ++kk) {    // 16 K-steps
      const int kBase = kk * 32;
      v16bf a0 = load_a_frag(sA, 0,  SA_STRIDE, kBase, lane);
      v16bf a1 = load_a_frag(sA, 16, SA_STRIDE, kBase, lane);
      const float* bp = wInE + (size_t)kBase * TWO_H + hGlob;
      v16bf bA = load_b_frag(bp, TWO_H, lane);
      v16bf bG = load_b_frag(bp + HDIM, TWO_H, lane);
      accA[0] = WMMA_BF16(a0, bA, accA[0]);
      accA[1] = WMMA_BF16(a1, bA, accA[1]);
      accG[0] = WMMA_BF16(a0, bG, accG[0]);
      accG[1] = WMMA_BF16(a1, bG, accG[1]);
    }

    // ---------------- GLU in C-register layout, spill bf16 chunk to LDS ----
    __syncthreads();   // previous chunk's GEMM2 reads of sH are done
#pragma unroll
    for (int i = 0; i < 2; ++i) {
#pragma unroll
      for (int v = 0; v < 8; ++v) {
        const float a = accA[i][v];
        const float g = accG[i][v];
        const float h = (a / (1.f + expf(-a))) * g;   // silu(a) * g
        const int mrow = i * 16 + v + ((lane >= 16) ? 8 : 0);
        sH[mrow * SH_STRIDE + hwBase + (lane & 15)] = (__bf16)h;
      }
    }
    __syncthreads();

    // ---------------- GEMM2 partial: C += [32 x 128] x [128 x 64] ----------
#pragma unroll
    for (int kk = 0; kk < SH_COLS / 32; ++kk) {   // 4 K-steps
      const int kBase = kk * 32;
      v16bf a0 = load_a_frag(sH, 0,  SH_STRIDE, kBase, lane);
      v16bf a1 = load_a_frag(sH, 16, SH_STRIDE, kBase, lane);
      const float* bro = wOutE + (size_t)(ch * SH_COLS + kBase) * DMODEL + nBase;
#pragma unroll
      for (int nt = 0; nt < 4; ++nt) {
        v16bf b = load_b_frag(bro + nt * 16, DMODEL, lane);
        accC[0][nt] = WMMA_BF16(a0, b, accC[0][nt]);
        accC[1][nt] = WMMA_BF16(a1, b, accC[1][nt]);
      }
    }
  }

  // ---- epilogue: scale rows by gate, scatter to ybuf[k][t][:] ----
#pragma unroll
  for (int i = 0; i < 2; ++i) {
#pragma unroll
    for (int v = 0; v < 8; ++v) {
      const int mrow = i * 16 + v + ((lane >= 16) ? 8 : 0);
      const int entry = sTok[mrow];
      if (entry < 0) continue;
      const int   t = entry >> 1;
      const int   k = entry & 1;
      const float g = sGate[mrow];
      float* yr = ybuf + ((size_t)k * T_TOK + t) * DMODEL + nBase + (lane & 15);
#pragma unroll
      for (int nt = 0; nt < 4; ++nt) yr[nt * 16] = g * accC[i][nt][v];
    }
  }
}

// ===========================================================================
// Combine: y = ybuf[0] + ybuf[1] + bias
// ===========================================================================
__global__ __launch_bounds__(256)
void combine_kernel(const float* __restrict__ ybuf,
                    const float* __restrict__ bias,
                    float* __restrict__ y) {
  const int idx = blockIdx.x * 256 + threadIdx.x;
  if (idx >= T_TOK * DMODEL) return;
  y[idx] = ybuf[idx] + ybuf[(size_t)T_TOK * DMODEL + idx] + bias[idx & (DMODEL - 1)];
}

// ===========================================================================
// Aux loss: deterministic fixed-order tree reductions.
// ===========================================================================
__global__ __launch_bounds__(256)
void loss_kernel(const float* __restrict__ probsBuf,
                 const float* __restrict__ lseSqBuf,
                 const int*   __restrict__ counts,
                 float* __restrict__ lossOut) {
  __shared__ float red[256];
  __shared__ float psumS[NEXP];
  const int tid = threadIdx.x;

  float acc[NEXP];
#pragma unroll
  for (int e = 0; e < NEXP; ++e) acc[e] = 0.f;
  for (int t = tid; t < T_TOK; t += 256)
#pragma unroll
    for (int e = 0; e < NEXP; ++e) acc[e] += probsBuf[t * NEXP + e];

  for (int e = 0; e < NEXP; ++e) {
    red[tid] = acc[e];
    __syncthreads();
    for (int off = 128; off > 0; off >>= 1) {
      if (tid < off) red[tid] += red[tid + off];
      __syncthreads();
    }
    if (tid == 0) psumS[e] = red[0];
    __syncthreads();
  }

  float lacc = 0.f;
  for (int t = tid; t < T_TOK; t += 256) lacc += lseSqBuf[t];
  red[tid] = lacc;
  __syncthreads();
  for (int off = 128; off > 0; off >>= 1) {
    if (tid < off) red[tid] += red[tid + off];
    __syncthreads();
  }

  if (tid == 0) {
    const float lsesq = red[0];
    float ps = 0.f, fs = 0.f;
    for (int e = 0; e < NEXP; ++e) { ps += psumS[e]; fs += (float)counts[e]; }
    float sw = 0.f;
    for (int e = 0; e < NEXP; ++e)
      sw += (psumS[e] / ps) * ((float)counts[e] / fs);
    sw *= (float)NEXP;
    lossOut[0] = sw + 0.1f * (lsesq / (float)T_TOK);
  }
}

// ===========================================================================
extern "C" void kernel_launch(void* const* d_in, const int* in_sizes, int n_in,
                              void* d_out, int out_size, void* d_ws, size_t ws_size,
                              hipStream_t stream) {
  (void)in_sizes; (void)n_in; (void)out_size; (void)ws_size;

  const float* x        = (const float*)d_in[0];
  const float* w_router = (const float*)d_in[1];
  const float* w_in     = (const float*)d_in[2];
  const float* w_out    = (const float*)d_in[3];
  const float* bias     = (const float*)d_in[4];

  // workspace carve-up (256B aligned)
  char*  ws  = (char*)d_ws;
  size_t off = 0;
  auto carve = [&](size_t bytes) -> char* {
    char* p = ws + off;
    off += (bytes + 255) & ~(size_t)255;
    return p;
  };
  int*   counts = (int*)  carve((size_t)NEXP * sizeof(int));
  int*   lists  = (int*)  carve((size_t)NEXP * T_TOK * sizeof(int));
  float* gatesW = (float*)carve((size_t)T_TOK * 2 * sizeof(float));
  float* probsB = (float*)carve((size_t)T_TOK * NEXP * sizeof(float));
  float* lseB   = (float*)carve((size_t)T_TOK * sizeof(float));
  float* ybuf   = (float*)carve((size_t)2 * T_TOK * DMODEL * sizeof(float));

  hipMemsetAsync(counts, 0, NEXP * sizeof(int), stream);

  router_kernel<<<T_TOK / 256, 256, 0, stream>>>(x, w_router, gatesW, counts,
                                                 lists, probsB, lseB);

  moe_ffn_kernel<<<NEXP * TILES_MAX, 256, 0, stream>>>(x, w_in, w_out, counts,
                                                       lists, gatesW, ybuf);

  combine_kernel<<<(T_TOK * DMODEL) / 256, 256, 0, stream>>>(ybuf, bias,
                                                             (float*)d_out);

  loss_kernel<<<1, 256, 0, stream>>>(probsB, lseB, counts,
                                     (float*)d_out + (size_t)T_TOK * DMODEL);
}
